// WarpFormerBlock_60619168415856
// MI455X (gfx1250) — matrix operand order repaired and checked by reading, not compiled, not run on hardware
//
#include <hip/hip_runtime.h>
#include <hip/hip_bf16.h>

typedef __attribute__((ext_vector_type(16))) _Float16 v16h;
typedef __attribute__((ext_vector_type(8)))  _Float16 v8h;
typedef __attribute__((ext_vector_type(8)))  float    v8f;

static constexpr int BB   = 2;
static constexpr int D    = 256;
static constexpr int HH   = 40;
static constexpr int WW   = 40;
static constexpr int NH   = 8;
static constexpr int HD   = 32;
static constexpr int HID  = 1024;
static constexpr int WSW  = 15;      // window side
static constexpr int W2S  = 225;     // window offsets
static constexpr int W2SP = 256;     // padded rel-weight rows (multiple of 64)
static constexpr int TOK  = BB * HH * WW;   // 3200
static constexpr float EPSF = 1e-5f;
static constexpr float NEGBIG = -3.0e38f;   // finite "-inf" (avoids inf-inf NaN)

// ---------------------------------------------------------------------------
// block-wide sum over 256 threads (8 wave32s)
// ---------------------------------------------------------------------------
__device__ __forceinline__ float blockSum256(float v) {
    __shared__ float sb[8];
    #pragma unroll
    for (int off = 16; off; off >>= 1) v += __shfl_xor(v, off, 32);
    if ((threadIdx.x & 31) == 0) sb[threadIdx.x >> 5] = v;
    __syncthreads();
    if (threadIdx.x == 0) {
        float t = 0.f;
        #pragma unroll
        for (int i = 0; i < 8; ++i) t += sb[i];
        sb[0] = t;
    }
    __syncthreads();
    float r = sb[0];
    __syncthreads();
    return r;
}

// ---------------------------------------------------------------------------
// f32 -> f16 weight conversion
// ---------------------------------------------------------------------------
__global__ void cvt_f16_kernel(const float* __restrict__ src,
                               _Float16* __restrict__ dst, int n) {
    int i = blockIdx.x * blockDim.x + threadIdx.x;
    if (i < n) dst[i] = (_Float16)src[i];
}

// Wrel [NH][225][32] f32 -> padded [NH][256][32] f16 (zero rows 225..255)
__global__ void cvt_rel_kernel(const float* __restrict__ src,
                               _Float16* __restrict__ dst) {
    int i = blockIdx.x * blockDim.x + threadIdx.x;
    if (i >= NH * W2SP * HD) return;
    int h = i / (W2SP * HD);
    int r = i - h * (W2SP * HD);
    int o = r / HD, c = r - o * HD;
    dst[i] = (o < W2S) ? (_Float16)src[(h * W2S + o) * HD + c] : (_Float16)0.f;
}

// ---------------------------------------------------------------------------
// Fused LN for q = LN(x), k = LN(xw), v = LN(xw+xm); copies x into f32 residual
// ---------------------------------------------------------------------------
__global__ __launch_bounds__(256)
void ln_qkv_kernel(const float* __restrict__ img, const float* __restrict__ imgw,
                   const float* __restrict__ maskv,
                   const float* __restrict__ g, const float* __restrict__ bln,
                   _Float16* __restrict__ q16, _Float16* __restrict__ k16,
                   _Float16* __restrict__ v16o, float* __restrict__ xres) {
    int t = blockIdx.x, c = threadIdx.x;
    int b = t & 1, p = t >> 1;
    long ii = ((long)(b * D + c)) * (HH * WW) + p;
    float x  = img[ii];
    float xw = imgw[ii];
    float v3 = xw + maskv[ii];

    float sx = blockSum256(x),  sxx = blockSum256(x * x);
    float sk = blockSum256(xw), skk = blockSum256(xw * xw);
    float sv = blockSum256(v3), svv = blockSum256(v3 * v3);

    const float inv = 1.f / (float)D;
    float mx = sx * inv, vx = sxx * inv - mx * mx;
    float mk = sk * inv, vk = skk * inv - mk * mk;
    float mv = sv * inv, vv = svv * inv - mv * mv;
    float gg = g[c], bb = bln[c];
    long oi = (long)t * D + c;
    q16[oi]  = (_Float16)((x  - mx) * rsqrtf(vx + EPSF) * gg + bb);
    k16[oi]  = (_Float16)((xw - mk) * rsqrtf(vk + EPSF) * gg + bb);
    v16o[oi] = (_Float16)((v3 - mv) * rsqrtf(vv + EPSF) * gg + bb);
    xres[oi] = x;
}

// LN over the f32 residual stream -> f16
__global__ __launch_bounds__(256)
void ln_kernel(const float* __restrict__ xin, const float* __restrict__ g,
               const float* __restrict__ bln, _Float16* __restrict__ out16) {
    int t = blockIdx.x, c = threadIdx.x;
    float x = xin[(long)t * D + c];
    float s = blockSum256(x), ss = blockSum256(x * x);
    const float inv = 1.f / (float)D;
    float mu = s * inv, var = ss * inv - mu * mu;
    out16[(long)t * D + c] =
        (_Float16)((x - mu) * rsqrtf(var + EPSF) * g[c] + bln[c]);
}

// ---------------------------------------------------------------------------
// Batched WMMA GEMM: C[m][n] (+)= scale * (sum_k A[m][k]*B[n][k] + bias[n])
// One wave computes a 32x64 C tile (2 A frags x 4 B frags -> 8 WMMAs/k-step).
// B must be padded to ceil(N/64)*64 rows (loads are unconditional).
// grid = (M/32, ceil(N/64), batch). Fragments use contiguous k = 16h + s in A and B.
// ---------------------------------------------------------------------------
__global__ __launch_bounds__(32)
void gemm_wmma_kernel(const _Float16* __restrict__ A, int ldA, long strideA,
                      const _Float16* __restrict__ Bw, int ldB, long strideB,
                      const float* __restrict__ bias, long strideBias,
                      float* __restrict__ Cf, _Float16* __restrict__ Ch,
                      int ldC, long strideC,
                      int N, int K, float outScale, int accumulate) {
    const int tm = blockIdx.x;
    const int tn = blockIdx.y;
    const int bz = blockIdx.z;
    const int lane  = threadIdx.x;
    const int half_ = lane >> 4;
    const int mr    = lane & 15;

    const _Float16* Ab = A  + strideA * bz;
    const _Float16* Bb = Bw + strideB * bz;
    const _Float16* aRow0 = Ab + (long)(tm * 32 + mr) * ldA;
    const _Float16* aRow1 = aRow0 + (long)16 * ldA;

    v8f acc[2][4] = {};

    for (int k0 = 0; k0 < K; k0 += 32) {
        v16h av0, av1;
        {   // A fragment K pattern: {16h..16h+15} contiguous
            v8h a0 = *(const v8h*)(aRow0 + k0 + 16 * half_);
            v8h a1 = *(const v8h*)(aRow0 + k0 + 8 + 16 * half_);
            v8h a2 = *(const v8h*)(aRow1 + k0 + 16 * half_);
            v8h a3 = *(const v8h*)(aRow1 + k0 + 8 + 16 * half_);
            #pragma unroll
            for (int i = 0; i < 8; ++i) {
                av0[i] = a0[i]; av0[i + 8] = a1[i];
                av1[i] = a2[i]; av1[i + 8] = a3[i];
            }
        }
        #pragma unroll
        for (int s = 0; s < 4; ++s) {
            int ng = tn * 64 + s * 16 + mr;       // B row (padded, always valid)
            v16h bv = *(const v16h*)(Bb + (long)ng * ldB + k0 + 16 * half_);
            acc[0][s] = __builtin_amdgcn_wmma_f32_16x16x32_f16(
                false, av0, false, bv, (short)0, acc[0][s], false, false);
            acc[1][s] = __builtin_amdgcn_wmma_f32_16x16x32_f16(
                false, av1, false, bv, (short)0, acc[1][s], false, false);
        }
    }

    #pragma unroll
    for (int s = 0; s < 4; ++s) {
        int ng = tn * 64 + s * 16 + mr;
        if (ng >= N) continue;
        float bvv = bias ? bias[strideBias * bz + ng] : 0.f;
        #pragma unroll
        for (int u = 0; u < 2; ++u) {
            #pragma unroll
            for (int r = 0; r < 8; ++r) {
                int m = tm * 32 + u * 16 + r + 8 * half_;
                float val = (acc[u][s][r] + bvv) * outScale;
                long ci = strideC * bz + (long)m * ldC + ng;
                if (Cf) { if (accumulate) Cf[ci] += val; else Cf[ci] = val; }
                if (Ch) Ch[ci] = (_Float16)val;
            }
        }
    }
}

// ---------------------------------------------------------------------------
// Windowed multi-head attention, WMMA formulation. One wave per (tile, head);
// tile = 16 queries along x at row y (x-tiles {0,16,32}; tile 2 clamped and
// store-guarded). For each window row dy, the 16 queries share a 32-wide key
// slab: QK = 2 WMMAs band-scattered into per-query 225-score LDS (rel bias
// pre-loaded, mask = -3e38); softmax per query; AV = 2 WMMAs with banded
// softmax weights as the A fragment and an LDS-transposed V slab as B.
// kp is pre-scaled by 1/sqrt(HD).
// ---------------------------------------------------------------------------
__global__ __launch_bounds__(32)
void attn_wmma_kernel(const _Float16* __restrict__ qp,
                      const _Float16* __restrict__ kp,
                      const _Float16* __restrict__ vp,
                      const float* __restrict__ rel,
                      _Float16* __restrict__ attnOut) {
    __shared__ float sS[16 * W2S];                    // [q][225]  14.4 KB
    __shared__ __align__(32) _Float16 sV[HD * 40];    // [d][key]   2.5 KB

    const int tile = blockIdx.x;          // b*120 + y*3 + xt
    const int h    = blockIdx.y;
    const int xt = tile % 3;
    const int y  = (tile / 3) % HH;
    const int b  = tile / (3 * HH);
    const int x0 = xt * 16;

    const int lane  = threadIdx.x;
    const int mr    = lane & 15;
    const int khalf = lane >> 4;

    // ---- Q fragment (A: 16 queries x K=32 head dims), clamped row index
    const int xq = (x0 + mr < WW) ? (x0 + mr) : (WW - 1);
    const _Float16* qrow = qp + (long)((y * WW + xq) * BB + b) * D + h * HD;
    v16h avQ;
    {
        v8h a0 = *(const v8h*)(qrow + 16 * khalf);
        v8h a1 = *(const v8h*)(qrow + 8 + 16 * khalf);
        #pragma unroll
        for (int i = 0; i < 8; ++i) { avQ[i] = a0[i]; avQ[i + 8] = a1[i]; }
    }

    // ---- init scores with rel bias; masked offsets -> -3e38
    for (int idx = lane; idx < 16 * W2S; idx += 32) {
        int m = idx / W2S, o = idx - m * W2S;
        int xm = (x0 + m < WW) ? (x0 + m) : (WW - 1);
        int dy = o / WSW, dx = o - dy * WSW;
        int yy = y + dy - 7, xx = xm + dx - 7;
        float v;
        if (yy >= 0 && yy < HH && xx >= 0 && xx < WW) {
            long tm_ = (long)((y * WW + xm) * BB + b);
            v = rel[tm_ * (NH * W2S) + h * W2S + o];
        } else {
            v = NEGBIG;
        }
        sS[idx] = v;
    }
    __syncthreads();

    // ---- QK: per window row, S = Q x K_slab, scatter along band dx = kk - m
    for (int d0 = 0; d0 < WSW; ++d0) {
        int yy = y + d0 - 7;
        if (yy < 0 || yy >= HH) continue;             // uniform per block
        #pragma unroll
        for (int j = 0; j < 2; ++j) {
            int kk = j * 16 + mr;                     // key slot / S column
            int xk = x0 - 7 + kk;
            v16h bv = {};
            if (xk >= 0 && xk < WW) {
                long tk = (long)((yy * WW + xk) * BB + b);
                bv = *(const v16h*)(kp + tk * D + h * HD + 16 * khalf);
            }
            v8f S = {};
            S = __builtin_amdgcn_wmma_f32_16x16x32_f16(
                false, avQ, false, bv, (short)0, S, false, false);
            #pragma unroll
            for (int r = 0; r < 8; ++r) {
                int m   = r + 8 * khalf;              // query row of S
                int dxv = kk - m;
                if (dxv >= 0 && dxv < WSW && xk >= 0 && xk < WW)
                    sS[m * W2S + d0 * WSW + dxv] += S[r];
            }
        }
    }
    __syncthreads();

    // ---- softmax per query over the 225 offsets
    for (int q = 0; q < 16; ++q) {
        float mx = NEGBIG;
        for (int o = lane; o < W2S; o += 32) mx = fmaxf(mx, sS[q * W2S + o]);
        #pragma unroll
        for (int off = 16; off; off >>= 1) mx = fmaxf(mx, __shfl_xor(mx, off, 32));
        float sum = 0.f;
        for (int o = lane; o < W2S; o += 32) {
            float e = expf(sS[q * W2S + o] - mx);     // masked -> 0
            sS[q * W2S + o] = e;
            sum += e;
        }
        #pragma unroll
        for (int off = 16; off; off >>= 1) sum += __shfl_xor(sum, off, 32);
        float rinv = 1.f / sum;
        for (int o = lane; o < W2S; o += 32) sS[q * W2S + o] *= rinv;
    }
    __syncthreads();

    // ---- AV: banded weights (A) x V slab (B), accumulate across window rows
    v8f c0 = {}, c1 = {};
    for (int d0 = 0; d0 < WSW; ++d0) {
        int yy = y + d0 - 7;
        if (yy < 0 || yy >= HH) continue;             // uniform per block
        // stage transposed V slab: lane = key slot
        {
            int xk = x0 - 7 + lane;
            if (xk >= 0 && xk < WW) {
                const _Float16* vr =
                    vp + (long)((yy * WW + xk) * BB + b) * D + h * HD;
                #pragma unroll
                for (int d = 0; d < HD; ++d) sV[d * 40 + lane] = vr[d];
            } else {
                #pragma unroll
                for (int d = 0; d < HD; ++d) sV[d * 40 + lane] = (_Float16)0.f;
            }
        }
        __syncthreads();
        // banded A fragment: A[m][key] = w[m][dy*15 + (key-m)] or 0
        v16h avW;
        #pragma unroll
        for (int jj = 0; jj < 16; ++jj) {
            int K   = jj + 16 * khalf;
            int dxv = K - mr;
            float wv = (dxv >= 0 && dxv < WSW)
                           ? sS[mr * W2S + d0 * WSW + dxv] : 0.f;
            avW[jj] = (_Float16)wv;
        }
        // B fragments: lane col d = j*16+mr, K halves contiguous in sV row d
        v16h bv0 = *(const v16h*)(sV + (0 * 16 + mr) * 40 + 16 * khalf);
        v16h bv1 = *(const v16h*)(sV + (1 * 16 + mr) * 40 + 16 * khalf);
        c0 = __builtin_amdgcn_wmma_f32_16x16x32_f16(
            false, avW, false, bv0, (short)0, c0, false, false);
        c1 = __builtin_amdgcn_wmma_f32_16x16x32_f16(
            false, avW, false, bv1, (short)0, c1, false, false);
        __syncthreads();
    }

    // ---- store (guard clamped tail queries)
    #pragma unroll
    for (int r = 0; r < 8; ++r) {
        int m = r + 8 * khalf;
        if (x0 + m < WW) {
            long to = (long)((y * WW + x0 + m) * BB + b);
            attnOut[to * D + h * HD + mr]      = (_Float16)c0[r];
            attnOut[to * D + h * HD + 16 + mr] = (_Float16)c1[r];
        }
    }
}

// ---------------------------------------------------------------------------
// GroupNorm stats over (channels-in-group x H x W) per (b, group)
// ---------------------------------------------------------------------------
__global__ __launch_bounds__(256)
void gn_stats_kernel(const float* __restrict__ m1, float* __restrict__ stats) {
    const int bg = blockIdx.x;
    const int b = bg >> 5, g = bg & 31;
    float s = 0.f, ss = 0.f;
    for (int i = threadIdx.x; i < 32 * HH * WW; i += 256) {
        int cl = i / (HH * WW);
        int p  = i - cl * (HH * WW);
        float v = m1[(long)(p * BB + b) * HID + g * 32 + cl];
        s += v; ss += v * v;
    }
    s = blockSum256(s); ss = blockSum256(ss);
    if (threadIdx.x == 0) {
        const float inv = 1.f / (32.f * HH * WW);
        float mean = s * inv;
        float var  = ss * inv - mean * mean;
        stats[bg * 2]     = mean;
        stats[bg * 2 + 1] = rsqrtf(var + EPSF);
    }
}

// GN apply + affine + exact GELU
__global__ void gn_gelu_kernel(const float* __restrict__ m1,
                               const float* __restrict__ stats,
                               const float* __restrict__ gg,
                               const float* __restrict__ gb,
                               float* __restrict__ act) {
    int i = blockIdx.x * blockDim.x + threadIdx.x;
    if (i >= TOK * HID) return;
    int t = i >> 10, c = i & 1023;
    int b = t & 1, grp = c >> 5;
    float mean = stats[(b * 32 + grp) * 2];
    float rstd = stats[(b * 32 + grp) * 2 + 1];
    float xv = (m1[i] - mean) * rstd * gg[c] + gb[c];
    act[i] = 0.5f * xv * (1.f + erff(xv * 0.70710678118654752f));
}

// 5x5 depthwise conv (pad 2, no bias), f32 in -> f16 out
__global__ void dwconv_kernel(const float* __restrict__ act,
                              const float* __restrict__ wdw,
                              _Float16* __restrict__ dw16) {
    int i = blockIdx.x * blockDim.x + threadIdx.x;
    if (i >= TOK * HID) return;
    int t = i >> 10, c = i & 1023;
    int b = t & 1, p = t >> 1;
    int y = p / WW, x = p - y * WW;
    float acc = 0.f;
    #pragma unroll
    for (int ky = 0; ky < 5; ++ky) {
        int yy = y + ky - 2;
        if (yy < 0 || yy >= HH) continue;
        #pragma unroll
        for (int kx = 0; kx < 5; ++kx) {
            int xx = x + kx - 2;
            if (xx < 0 || xx >= WW) continue;
            acc += wdw[c * 25 + ky * 5 + kx] *
                   act[(long)((yy * WW + xx) * BB + b) * HID + c];
        }
    }
    dw16[i] = (_Float16)acc;
}

// token-major residual -> [B, C, H, W] output
__global__ void out_kernel(const float* __restrict__ xres, float* __restrict__ outp) {
    int i = blockIdx.x * blockDim.x + threadIdx.x;
    if (i >= BB * D * HH * WW) return;
    int b = i / (D * HH * WW);
    int r = i - b * (D * HH * WW);
    int c = r / (HH * WW);
    int p = r - c * (HH * WW);
    outp[i] = xres[(long)(p * BB + b) * D + c];
}

// ---------------------------------------------------------------------------
extern "C" void kernel_launch(void* const* d_in, const int* in_sizes, int n_in,
                              void* d_out, int out_size, void* d_ws, size_t ws_size,
                              hipStream_t stream) {
    (void)in_sizes; (void)n_in; (void)out_size; (void)ws_size;
    const float* img   = (const float*)d_in[0];
    const float* imgw  = (const float*)d_in[1];
    const float* maskv = (const float*)d_in[2];
    const float* ln1_g = (const float*)d_in[3];
    const float* ln1_b = (const float*)d_in[4];
    const float* ln2_g = (const float*)d_in[5];
    const float* ln2_b = (const float*)d_in[6];
    const float* Wq    = (const float*)d_in[7];
    const float* bq    = (const float*)d_in[8];
    const float* Wk    = (const float*)d_in[9];
    const float* bk    = (const float*)d_in[10];
    const float* Wv    = (const float*)d_in[11];
    const float* bv    = (const float*)d_in[12];
    const float* Wrel  = (const float*)d_in[13];
    const float* brel  = (const float*)d_in[14];
    const float* Wproj = (const float*)d_in[15];
    const float* bproj = (const float*)d_in[16];
    const float* W1    = (const float*)d_in[17];
    const float* bW1   = (const float*)d_in[18];
    const float* gn_g  = (const float*)d_in[19];
    const float* gn_b  = (const float*)d_in[20];
    const float* Wdw   = (const float*)d_in[21];
    const float* W2    = (const float*)d_in[22];
    const float* bW2   = (const float*)d_in[23];

    char* ws = (char*)d_ws;
    size_t off = 0;
    auto alloc = [&](size_t bytes) -> void* {
        off = (off + 255) & ~((size_t)255);
        void* p = ws + off;
        off += bytes;
        return p;
    };

    // f16 weights (B operands padded to multiples of 64 rows)
    _Float16* wq16    = (_Float16*)alloc((size_t)D * D * 2);
    _Float16* wk16    = (_Float16*)alloc((size_t)D * D * 2);
    _Float16* wv16    = (_Float16*)alloc((size_t)D * D * 2);
    _Float16* wproj16 = (_Float16*)alloc((size_t)D * D * 2);
    _Float16* wrel16  = (_Float16*)alloc((size_t)NH * W2SP * HD * 2);
    _Float16* w116    = (_Float16*)alloc((size_t)HID * D * 2);
    _Float16* w216    = (_Float16*)alloc((size_t)D * HID * 2);
    // activations
    _Float16* q16    = (_Float16*)alloc((size_t)TOK * D * 2);
    _Float16* k16    = (_Float16*)alloc((size_t)TOK * D * 2);
    _Float16* v16b   = (_Float16*)alloc((size_t)TOK * D * 2);
    _Float16* qp16   = (_Float16*)alloc((size_t)TOK * D * 2);
    _Float16* kp16   = (_Float16*)alloc((size_t)TOK * D * 2);
    _Float16* vp16   = (_Float16*)alloc((size_t)TOK * D * 2);
    _Float16* attn16 = (_Float16*)alloc((size_t)TOK * D * 2);
    _Float16* h16    = (_Float16*)alloc((size_t)TOK * D * 2);
    _Float16* dw16   = (_Float16*)alloc((size_t)TOK * HID * 2);
    float* xres  = (float*)alloc((size_t)TOK * D * 4);
    float* rel   = (float*)alloc((size_t)TOK * NH * W2S * 4);
    float* m1    = (float*)alloc((size_t)TOK * HID * 4);
    float* act   = (float*)alloc((size_t)TOK * HID * 4);
    float* gstat = (float*)alloc((size_t)BB * 32 * 2 * 4);

    auto cvt = [&](const float* src, _Float16* dst, int n) {
        cvt_f16_kernel<<<(n + 255) / 256, 256, 0, stream>>>(src, dst, n);
    };
    cvt(Wq, wq16, D * D);
    cvt(Wk, wk16, D * D);
    cvt(Wv, wv16, D * D);
    cvt(Wproj, wproj16, D * D);
    cvt(W1, w116, HID * D);
    cvt(W2, w216, D * HID);
    cvt_rel_kernel<<<(NH * W2SP * HD + 255) / 256, 256, 0, stream>>>(Wrel, wrel16);

    // LN1 (q/k/v) + residual copy
    ln_qkv_kernel<<<TOK, 256, 0, stream>>>(img, imgw, maskv, ln1_g, ln1_b,
                                           q16, k16, v16b, xres);

    const float kscale = 0.1767766952966369f;  // 1/sqrt(HD), folded into K proj
    const int MT = TOK / 32;                   // 100 row tiles (32-wide)

    // Q/K/V projections (WMMA)
    gemm_wmma_kernel<<<dim3(MT, 4, 1), 32, 0, stream>>>(
        q16, D, 0, wq16, D, 0, bq, 0, nullptr, qp16, D, 0, D, D, 1.f, 0);
    gemm_wmma_kernel<<<dim3(MT, 4, 1), 32, 0, stream>>>(
        k16, D, 0, wk16, D, 0, bk, 0, nullptr, kp16, D, 0, D, D, kscale, 0);
    gemm_wmma_kernel<<<dim3(MT, 4, 1), 32, 0, stream>>>(
        v16b, D, 0, wv16, D, 0, bv, 0, nullptr, vp16, D, 0, D, D, 1.f, 0);

    // q-conditioned relative bias: per-head [TOK x 32] @ [32 x 225]^T
    gemm_wmma_kernel<<<dim3(MT, 4, NH), 32, 0, stream>>>(
        qp16, D, HD, wrel16, HD, (long)W2SP * HD, brel, W2S,
        rel, nullptr, NH * W2S, W2S, W2S, HD, 1.f, 0);

    // windowed attention core (WMMA QK + AV)
    attn_wmma_kernel<<<dim3(BB * HH * 3, NH), 32, 0, stream>>>(
        qp16, kp16, vp16, rel, attn16);

    // output projection, accumulate into residual
    gemm_wmma_kernel<<<dim3(MT, 4, 1), 32, 0, stream>>>(
        attn16, D, 0, wproj16, D, 0, bproj, 0, xres, nullptr, D, 0, D, D, 1.f, 1);

    // LN2 -> W1 -> GN/GELU -> dwconv -> W2 (accumulate into residual)
    ln_kernel<<<TOK, 256, 0, stream>>>(xres, ln2_g, ln2_b, h16);
    gemm_wmma_kernel<<<dim3(MT, 16, 1), 32, 0, stream>>>(
        h16, D, 0, w116, D, 0, bW1, 0, m1, nullptr, HID, 0, HID, D, 1.f, 0);
    gn_stats_kernel<<<BB * 32, 256, 0, stream>>>(m1, gstat);
    gn_gelu_kernel<<<(TOK * HID + 255) / 256, 256, 0, stream>>>(m1, gstat, gn_g,
                                                                gn_b, act);
    dwconv_kernel<<<(TOK * HID + 255) / 256, 256, 0, stream>>>(act, Wdw, dw16);
    gemm_wmma_kernel<<<dim3(MT, 4, 1), 32, 0, stream>>>(
        dw16, HID, 0, w216, HID, 0, bW2, 0, xres, nullptr, D, 0, D, HID, 1.f, 1);

    // back to [B, C, H, W]
    out_kernel<<<(BB * D * HH * WW + 255) / 256, 256, 0, stream>>>(
        xres, (float*)d_out);
}
